// SequenceAtomTransformerBlock_18597208392251
// MI455X (gfx1250) — matrix operand order repaired and checked by reading, not compiled, not run on hardware
//
#include <hip/hip_runtime.h>
#include <hip/hip_bf16.h>
#include <math.h>

#define B_  2
#define N_  16384
#define R_  2048
#define CA_ 128
#define CS_ 256
#define CP_ 16
#define H_  4
#define WQ_ 32
#define WK_ 128
#define NW_ (N_ / WQ_)   // 512
#define CH_ (CA_ / H_)   // 32

typedef __attribute__((ext_vector_type(16))) __bf16 v16bf;
typedef __attribute__((ext_vector_type(8)))  float  v8f;

// ---------------------------------------------------------------------------
// WMMA helpers (CDNA5 wave32, 16x16x32 bf16 -> f32)
// ---------------------------------------------------------------------------
__device__ __forceinline__ v8f wmma_bf16(v16bf a, v16bf b, v8f c) {
  // (neg_a, A, neg_b, B, c_mod, C, reuse_a, reuse_b)
  return __builtin_amdgcn_wmma_f32_16x16x32_bf16(false, a, false, b, (short)0, c,
                                                 false, false);
}

// A fragment (16x32 bf16), row-major LDS, stride lda elements, K-chunk at k0.
// ISA layout: lane m=lane&15; half=lane>>4; elems 0..7 -> K=k0+half*8+j,
// elems 8..15 -> K=k0+16+half*8+j. Both runs are 16B-contiguous -> ds_load_b128.
__device__ __forceinline__ v16bf fragA_lds(const __bf16* A, int lda, int k0) {
  const int lane = threadIdx.x & 31;
  const int m = lane & 15, half = lane >> 4;
  const __bf16* p = A + m * lda + k0 + half * 8;
  v16bf f;
#pragma unroll
  for (int j = 0; j < 8; ++j) { f[j] = p[j]; f[j + 8] = p[j + 16]; }
  return f;
}

// B fragment (32x16 bf16) from a transposed (N-major) image Bt[n][k]:
// lane n=lane&15, half=lane>>4; elems 0..7 -> K = k0 + half*8 + j, 8..15 -> K = k0 + 16 + half*8 + j.
// Works for LDS or global bf16 images (merged into b128 loads either way).
__device__ __forceinline__ v16bf fragB_bt(const __bf16* Bt, int ldk, int k0,
                                          int n0) {
  const int lane = threadIdx.x & 31;
  const int n = n0 + (lane & 15), half = lane >> 4;
  const __bf16* p = Bt + (size_t)n * ldk + k0 + half * 8;
  v16bf f;
#pragma unroll
  for (int j = 0; j < 8; ++j) { f[j] = p[j]; f[j + 8] = p[j + 16]; }
  return f;
}

__device__ __forceinline__ float sigmoidf_(float x) {
  return 1.0f / (1.0f + __expf(-x));
}

// ---------------------------------------------------------------------------
// Kernel 0: one-time weight prep. src: row-major f32 (K x Nn).
// dst: bf16 "Bt" image, N-major (Nn x K) so B-fragments are contiguous.
// ---------------------------------------------------------------------------
__global__ __launch_bounds__(256)
void k0_wprep(const float* __restrict__ src, __bf16* __restrict__ dst, int K,
              int Nn) {
  const int i = blockIdx.x * 256 + threadIdx.x;
  if (i < K * Nn) {
    const int k = i / Nn, n = i % Nn;
    dst[(size_t)n * K + k] = (__bf16)src[i];
  }
}

// ---------------------------------------------------------------------------
// Kernel 1: per-residue conditioning (6 fused 256->128 WMMA GEMMs).
// ---------------------------------------------------------------------------
__global__ __launch_bounds__(128)
void k1_res_cond(const float* __restrict__ res,
                 const float* __restrict__ a_ln_w, const float* __restrict__ t_ln_w,
                 const __bf16* __restrict__ wgA, const __bf16* __restrict__ wbA,
                 const __bf16* __restrict__ wgT, const __bf16* __restrict__ wbT,
                 const __bf16* __restrict__ wsg, const __bf16* __restrict__ wtc,
                 const float* __restrict__ a_bg, const float* __restrict__ t_bg,
                 const float* __restrict__ sg_b, const float* __restrict__ t_bc,
                 float* __restrict__ gateA, float* __restrict__ biasA,
                 float* __restrict__ gateT, float* __restrict__ biasT,
                 float* __restrict__ gS,    float* __restrict__ gT) {
  __shared__ __bf16 cA[16][CS_];
  __shared__ __bf16 cT[16][CS_];
  __shared__ __bf16 rw[16][CS_];

  const int t = threadIdx.x;
  const int row = t >> 3;  // 16 rows, 8 threads each
  const int sub = t & 7;
  const size_t grow = (size_t)blockIdx.x * 16 + row;
  const float* rp = res + grow * CS_;

  float sum = 0.f, sq = 0.f;
  float vals[32];
#pragma unroll
  for (int j = 0; j < 32; ++j) {
    float v = rp[sub * 32 + j];
    vals[j] = v; sum += v; sq += v * v;
  }
  sum += __shfl_xor(sum, 1); sum += __shfl_xor(sum, 2); sum += __shfl_xor(sum, 4);
  sq  += __shfl_xor(sq, 1);  sq  += __shfl_xor(sq, 2);  sq  += __shfl_xor(sq, 4);
  const float mean = sum * (1.0f / CS_);
  const float var  = sq * (1.0f / CS_) - mean * mean;
  const float rstd = rsqrtf(var + 1e-5f);
#pragma unroll
  for (int j = 0; j < 32; ++j) {
    const int c = sub * 32 + j;
    const float ln = (vals[j] - mean) * rstd;
    cA[row][c] = (__bf16)(ln * a_ln_w[c]);
    cT[row][c] = (__bf16)(ln * t_ln_w[c]);
    rw[row][c] = (__bf16)vals[j];
  }
  __syncthreads();

  const int wave = t >> 5, lane = t & 31;
  const int lcol = lane & 15, half = lane >> 4;
  const size_t base = (size_t)blockIdx.x * 16;

  // Literal pointers per call -> keeps global/LDS address-space provenance.
  auto gemm_one = [&](const __bf16* Alds, const __bf16* Wbt, const float* bias,
                      bool dosig, float* outp) __attribute__((always_inline)) {
#pragma unroll
    for (int nt = 0; nt < 2; ++nt) {
      const int n0 = (wave * 2 + nt) * 16;
      v8f acc = {};
#pragma unroll
      for (int kc = 0; kc < CS_ / 32; ++kc) {
        v16bf a = fragA_lds(Alds, CS_, kc * 32);
        v16bf b = fragB_bt(Wbt, CS_, kc * 32, n0);
        acc = wmma_bf16(a, b, acc);
      }
      const float bv = bias ? bias[n0 + lcol] : 0.f;
#pragma unroll
      for (int r = 0; r < 8; ++r) {
        const int rr = r + half * 8;
        float v = acc[r] + bv;
        if (dosig) v = sigmoidf_(v);
        outp[(base + rr) * CA_ + n0 + lcol] = v;
      }
    }
  };
  gemm_one(&cA[0][0], wgA, a_bg, true, gateA);
  gemm_one(&cA[0][0], wbA, nullptr, false, biasA);
  gemm_one(&cT[0][0], wgT, t_bg, true, gateT);
  gemm_one(&cT[0][0], wbT, nullptr, false, biasT);
  gemm_one(&rw[0][0], wsg, sg_b, true, gS);
  gemm_one(&rw[0][0], wtc, t_bc, true, gT);
}

// ---------------------------------------------------------------------------
// Kernel 2: per-atom AdaLN + q/kv projections (bf16 outputs for attention).
// ---------------------------------------------------------------------------
__global__ __launch_bounds__(128)
void k2_atom_qkv(const float* __restrict__ atomF, const int* __restrict__ ari,
                 const float* __restrict__ gateA, const float* __restrict__ biasA,
                 const __bf16* __restrict__ wq, const float* __restrict__ q_b,
                 const __bf16* __restrict__ wkv,
                 __bf16* __restrict__ qWs, __bf16* __restrict__ kvWs) {
  __shared__ __bf16 sL[16][CA_];
  const int t = threadIdx.x;
  const int row = t >> 3, sub = t & 7;
  const size_t grow = (size_t)blockIdx.x * 16 + row;
  const int b = (int)(grow / N_);
  const int idx = ari[grow];
  const float* xp = atomF + grow * CA_;

  float sum = 0.f, sq = 0.f;
  float vals[16];
#pragma unroll
  for (int j = 0; j < 16; ++j) {
    float v = xp[sub * 16 + j];
    vals[j] = v; sum += v; sq += v * v;
  }
  sum += __shfl_xor(sum, 1); sum += __shfl_xor(sum, 2); sum += __shfl_xor(sum, 4);
  sq  += __shfl_xor(sq, 1);  sq  += __shfl_xor(sq, 2);  sq  += __shfl_xor(sq, 4);
  const float mean = sum * (1.0f / CA_);
  const float var  = sq * (1.0f / CA_) - mean * mean;
  const float rstd = rsqrtf(var + 1e-5f);
  const float* gp = gateA + ((size_t)b * R_ + idx) * CA_;
  const float* bp = biasA + ((size_t)b * R_ + idx) * CA_;
#pragma unroll
  for (int j = 0; j < 16; ++j) {
    const int c = sub * 16 + j;
    const float ln = (vals[j] - mean) * rstd;
    sL[row][c] = (__bf16)(ln * gp[c] + bp[c]);  // gate already sigmoided
  }
  __syncthreads();

  const int wave = t >> 5, lane = t & 31;
  const int lcol = lane & 15, half = lane >> 4;
  const size_t base = (size_t)blockIdx.x * 16;

  // q = s @ q_W + q_b
#pragma unroll
  for (int nt = 0; nt < 2; ++nt) {
    const int n0 = (wave * 2 + nt) * 16;
    v8f acc = {};
#pragma unroll
    for (int kc = 0; kc < 4; ++kc) {
      v16bf a = fragA_lds(&sL[0][0], CA_, kc * 32);
      v16bf bw = fragB_bt(wq, CA_, kc * 32, n0);
      acc = wmma_bf16(a, bw, acc);
    }
    const float qb = q_b[n0 + lcol];
#pragma unroll
    for (int r = 0; r < 8; ++r) {
      const int rr = r + half * 8;
      qWs[(base + rr) * CA_ + n0 + lcol] = (__bf16)(acc[r] + qb);
    }
  }
  // kv = s @ kv_W
#pragma unroll
  for (int nt = 0; nt < 4; ++nt) {
    const int n0 = (wave * 4 + nt) * 16;
    v8f acc = {};
#pragma unroll
    for (int kc = 0; kc < 4; ++kc) {
      v16bf a = fragA_lds(&sL[0][0], CA_, kc * 32);
      v16bf bw = fragB_bt(wkv, CA_, kc * 32, n0);
      acc = wmma_bf16(a, bw, acc);
    }
#pragma unroll
    for (int r = 0; r < 8; ++r) {
      const int rr = r + half * 8;
      kvWs[(base + rr) * (2 * CA_) + n0 + lcol] = (__bf16)acc[r];
    }
  }
}

// ---------------------------------------------------------------------------
// Kernel 3: windowed attention. One block per (b, window). 8 waves:
// wave = head(4) x m-tile(2). Streams atompair once for the pair bias.
// ---------------------------------------------------------------------------
__global__ __launch_bounds__(256)
void k3_attn(const __bf16* __restrict__ qWs, const __bf16* __restrict__ kvWs,
             const float* __restrict__ ap, const float* __restrict__ bij_ln_w,
             const float* __restrict__ bij_ln_b, const float* __restrict__ bij_W,
             const int* __restrict__ ari, const float* __restrict__ gS,
             const float* __restrict__ amask, const float* __restrict__ atomF,
             float* __restrict__ atomInter) {
  extern __shared__ char smem[];
  __bf16* qL  = (__bf16*)smem;            // 32 x 128
  __bf16* kL  = qL + 32 * 128;            // 128 keys x 128 feats (row-major)
  __bf16* vT  = kL + 128 * 128;           // 128 feats x 128 keys (transposed)
  float*  bij = (float*)(vT + 128 * 128); // [h*32+qi][kj] f32
  __bf16* pL  = (__bf16*)(bij + H_ * WQ_ * WK_);  // 8 waves x 16 x 128

  const int blk = blockIdx.x;
  const int b = blk / NW_, nw = blk % NW_;
  const int t = threadIdx.x;

  // ---- phase A: cooperative staging (contiguous per-thread -> b128 merges) --
  {
    const size_t qbase = ((size_t)b * N_ + (size_t)nw * WQ_) * CA_;
    const int i0 = t * 16;  // 256 threads x 16 = 4096
#pragma unroll
    for (int j = 0; j < 16; ++j) qL[i0 + j] = qWs[qbase + i0 + j];
  }
  {
    const int key = t >> 1;            // 128 keys, 2 threads/key
    const int c0 = (t & 1) * 64;
    int kid = nw * WQ_ - 48 + key;
    kid = kid < 0 ? 0 : (kid > N_ - 1 ? N_ - 1 : kid);
    const __bf16* src = kvWs + ((size_t)b * N_ + kid) * (2 * CA_);
#pragma unroll
    for (int j = 0; j < 64; ++j) {
      kL[key * CA_ + c0 + j] = src[c0 + j];           // contiguous load+store
      vT[(c0 + j) * WK_ + key] = src[CA_ + c0 + j];   // contiguous load
    }
  }
  {
    float blw[CP_], blb[CP_], bw[CP_][H_];
#pragma unroll
    for (int e = 0; e < CP_; ++e) {
      blw[e] = bij_ln_w[e]; blb[e] = bij_ln_b[e];
#pragma unroll
      for (int h = 0; h < H_; ++h) bw[e][h] = bij_W[e * H_ + h];
    }
    for (int pi = t; pi < WQ_ * WK_; pi += 256) {
      const int qi = pi >> 7, kj = pi & 127;
      const float* pp =
          ap + ((((size_t)b * NW_ + nw) * WQ_ + qi) * WK_ + kj) * CP_;
      __builtin_prefetch(pp + 256 * CP_, 0, 1);  // stream-ahead hint
      float ev[CP_], s = 0.f, q2 = 0.f;
#pragma unroll
      for (int e = 0; e < CP_; ++e) {
        const float v = pp[e];
        ev[e] = v; s += v; q2 += v * v;
      }
      const float mean = s * (1.0f / CP_);
      const float var  = q2 * (1.0f / CP_) - mean * mean;
      const float rstd = rsqrtf(var + 1e-5f);
      float acc[H_] = {0.f, 0.f, 0.f, 0.f};
#pragma unroll
      for (int e = 0; e < CP_; ++e) {
        const float ln = (ev[e] - mean) * rstd * blw[e] + blb[e];
#pragma unroll
        for (int h = 0; h < H_; ++h) acc[h] += ln * bw[e][h];
      }
#pragma unroll
      for (int h = 0; h < H_; ++h) bij[(h * WQ_ + qi) * WK_ + kj] = acc[h];
    }
  }
  __syncthreads();

  // ---- phase B: scores + softmax (per wave: one head, one 16-row m-tile) ----
  const int wave = t >> 5, lane = t & 31;
  const int lcol = lane & 15, half = lane >> 4;
  const int h = wave & 3, mt = wave >> 2;
  const float scale = 0.1767766952966369f;  // 1/sqrt(32)

  const v16bf aq = fragA_lds(qL + (mt * 16) * CA_, CA_, h * CH_);
  float S[8][8];
#pragma unroll
  for (int nt = 0; nt < 8; ++nt) {
    v16bf bk = fragB_bt(kL, CA_, h * CH_, nt * 16);
    v8f acc = {};
    acc = wmma_bf16(aq, bk, acc);
    const int kj = nt * 16 + lcol;
    const int kid = nw * WQ_ - 48 + kj;
    const float pen = (kid >= 0 && kid < N_) ? 0.f : -1.0e8f;
#pragma unroll
    for (int r = 0; r < 8; ++r) {
      const int qi = mt * 16 + r + half * 8;
      S[nt][r] = acc[r] * scale + bij[(h * WQ_ + qi) * WK_ + kj] + pen;
    }
  }
#pragma unroll
  for (int r = 0; r < 8; ++r) {
    float m = S[0][r];
#pragma unroll
    for (int nt = 1; nt < 8; ++nt) m = fmaxf(m, S[nt][r]);
    m = fmaxf(m, __shfl_xor(m, 1)); m = fmaxf(m, __shfl_xor(m, 2));
    m = fmaxf(m, __shfl_xor(m, 4)); m = fmaxf(m, __shfl_xor(m, 8));
    float s = 0.f;
#pragma unroll
    for (int nt = 0; nt < 8; ++nt) {
      const float e = __expf(S[nt][r] - m);
      S[nt][r] = e; s += e;
    }
    s += __shfl_xor(s, 1); s += __shfl_xor(s, 2);
    s += __shfl_xor(s, 4); s += __shfl_xor(s, 8);
    const float inv = 1.0f / s;
#pragma unroll
    for (int nt = 0; nt < 8; ++nt) S[nt][r] *= inv;
  }
  __bf16* pw = pL + wave * 16 * WK_;
#pragma unroll
  for (int nt = 0; nt < 8; ++nt)
#pragma unroll
    for (int r = 0; r < 8; ++r)
      pw[(r + half * 8) * WK_ + nt * 16 + lcol] = (__bf16)S[nt][r];
  __syncthreads();

  // ---- phase C: O = P @ V_h, fused gate/mask/residual ----
  const v8f vzero = {};
  v8f acc2[2] = {vzero, vzero};
#pragma unroll
  for (int kc = 0; kc < 4; ++kc) {
    v16bf aP = fragA_lds(pw, WK_, kc * 32);
#pragma unroll
    for (int nt2 = 0; nt2 < 2; ++nt2) {
      v16bf bV = fragB_bt(vT, WK_, kc * 32, h * CH_ + nt2 * 16);
      acc2[nt2] = wmma_bf16(aP, bV, acc2[nt2]);
    }
  }
#pragma unroll
  for (int nt2 = 0; nt2 < 2; ++nt2)
#pragma unroll
    for (int r = 0; r < 8; ++r) {
      const int lrow = mt * 16 + r + half * 8;
      const size_t a = (size_t)b * N_ + (size_t)nw * WQ_ + lrow;
      const int c = h * CH_ + nt2 * 16 + lcol;
      const int idx = ari[a];
      float v = acc2[nt2][r] * gS[((size_t)b * R_ + idx) * CA_ + c];
      v = v * amask[a] + atomF[a * CA_ + c];
      atomInter[a * CA_ + c] = v;
    }
}

// ---------------------------------------------------------------------------
// Kernel 4: gated SwiGLU transition + final residual -> d_out.
// ---------------------------------------------------------------------------
__global__ __launch_bounds__(128)
void k4_transition(const float* __restrict__ atomInter, const int* __restrict__ ari,
                   const float* __restrict__ amask,
                   const float* __restrict__ gateT, const float* __restrict__ biasT,
                   const float* __restrict__ gT,
                   const __bf16* __restrict__ w1, const __bf16* __restrict__ w2,
                   const __bf16* __restrict__ wo, float* __restrict__ out) {
  __shared__ __bf16 tL[16][CA_];
  __shared__ __bf16 bL[16][2 * CA_];
  const int t = threadIdx.x;
  const int row = t >> 3, sub = t & 7;
  const size_t grow = (size_t)blockIdx.x * 16 + row;
  const int b = (int)(grow / N_);
  const int idx = ari[grow];
  const float* xp = atomInter + grow * CA_;

  float sum = 0.f, sq = 0.f;
  float vals[16];
#pragma unroll
  for (int j = 0; j < 16; ++j) {
    float v = xp[sub * 16 + j];
    vals[j] = v; sum += v; sq += v * v;
  }
  sum += __shfl_xor(sum, 1); sum += __shfl_xor(sum, 2); sum += __shfl_xor(sum, 4);
  sq  += __shfl_xor(sq, 1);  sq  += __shfl_xor(sq, 2);  sq  += __shfl_xor(sq, 4);
  const float mean = sum * (1.0f / CA_);
  const float var  = sq * (1.0f / CA_) - mean * mean;
  const float rstd = rsqrtf(var + 1e-5f);
  const float* gp = gateT + ((size_t)b * R_ + idx) * CA_;
  const float* bp = biasT + ((size_t)b * R_ + idx) * CA_;
#pragma unroll
  for (int j = 0; j < 16; ++j) {
    const int c = sub * 16 + j;
    const float ln = (vals[j] - mean) * rstd;
    tL[row][c] = (__bf16)(ln * gp[c] + bp[c]);
  }
  __syncthreads();

  const int wave = t >> 5, lane = t & 31;
  const int lcol = lane & 15, half = lane >> 4;
  const size_t base = (size_t)blockIdx.x * 16;

  // h1 = t@W1 ; h2 = t@W2 ; bvec = silu(h1)*h2
#pragma unroll
  for (int nt = 0; nt < 4; ++nt) {
    const int n0 = (wave * 4 + nt) * 16;
    v8f acc1 = {}, acc2 = {};
#pragma unroll
    for (int kc = 0; kc < 4; ++kc) {
      v16bf a = fragA_lds(&tL[0][0], CA_, kc * 32);
      v16bf b1 = fragB_bt(w1, CA_, kc * 32, n0);
      acc1 = wmma_bf16(a, b1, acc1);
      v16bf b2 = fragB_bt(w2, CA_, kc * 32, n0);
      acc2 = wmma_bf16(a, b2, acc2);
    }
#pragma unroll
    for (int r = 0; r < 8; ++r) {
      const int rr = r + half * 8;
      const float x = acc1[r];
      bL[rr][n0 + lcol] = (__bf16)((x * sigmoidf_(x)) * acc2[r]);
    }
  }
  __syncthreads();

  // trans = tg * (bvec @ t_Wo) ; out = atom + trans*mask
#pragma unroll
  for (int nt2 = 0; nt2 < 2; ++nt2) {
    const int n0 = (wave * 2 + nt2) * 16;
    v8f acc = {};
#pragma unroll
    for (int kc = 0; kc < 8; ++kc) {
      v16bf a = fragA_lds(&bL[0][0], 2 * CA_, kc * 32);
      v16bf bw = fragB_bt(wo, 2 * CA_, kc * 32, n0);
      acc = wmma_bf16(a, bw, acc);
    }
#pragma unroll
    for (int r = 0; r < 8; ++r) {
      const int rr = r + half * 8;
      const size_t g = base + rr;
      const int bb = (int)(g / N_);
      const int ii = ari[g];
      const int c = n0 + lcol;
      const float tg = gT[((size_t)bb * R_ + ii) * CA_ + c];
      out[g * CA_ + c] = atomInter[g * CA_ + c] + tg * acc[r] * amask[g];
    }
  }
}

// ---------------------------------------------------------------------------
extern "C" void kernel_launch(void* const* d_in, const int* in_sizes, int n_in,
                              void* d_out, int out_size, void* d_ws, size_t ws_size,
                              hipStream_t stream) {
  (void)in_sizes; (void)n_in; (void)out_size; (void)ws_size;
  const float* atomF    = (const float*)d_in[0];
  const float* resF     = (const float*)d_in[1];
  const float* apF      = (const float*)d_in[2];
  const int*   ari      = (const int*)d_in[3];
  const float* amask    = (const float*)d_in[4];
  // d_in[5] atompair_mask: recomputed analytically inside k3 (clip window)
  const float* a_ln_w   = (const float*)d_in[6];
  const float* a_Wg     = (const float*)d_in[7];
  const float* a_bg     = (const float*)d_in[8];
  const float* a_Wb     = (const float*)d_in[9];
  const float* q_W      = (const float*)d_in[10];
  const float* q_b      = (const float*)d_in[11];
  const float* kv_W     = (const float*)d_in[12];
  const float* bij_ln_w = (const float*)d_in[13];
  const float* bij_ln_b = (const float*)d_in[14];
  const float* bij_W    = (const float*)d_in[15];
  const float* sg_W     = (const float*)d_in[16];
  const float* sg_b     = (const float*)d_in[17];
  const float* t_ln_w   = (const float*)d_in[18];
  const float* t_Wg     = (const float*)d_in[19];
  const float* t_bg     = (const float*)d_in[20];
  const float* t_Wb     = (const float*)d_in[21];
  const float* t_W1     = (const float*)d_in[22];
  const float* t_W2     = (const float*)d_in[23];
  const float* t_Wc     = (const float*)d_in[24];
  const float* t_bc     = (const float*)d_in[25];
  const float* t_Wo     = (const float*)d_in[26];
  float* out = (float*)d_out;

  const size_t BRCA = (size_t)B_ * R_ * CA_;
  const size_t BNCA = (size_t)B_ * N_ * CA_;
  float* gateA = (float*)d_ws;
  float* biasA = gateA + BRCA;
  float* gateT = biasA + BRCA;
  float* biasT = gateT + BRCA;
  float* gS    = biasT + BRCA;
  float* gT    = gS + BRCA;
  __bf16* qWs  = (__bf16*)(gT + BRCA);
  __bf16* kvWs = qWs + BNCA;
  float* atomInter = (float*)(kvWs + 2 * BNCA);
  // bf16 N-major weight images
  __bf16* wimg = (__bf16*)(atomInter + BNCA);
  __bf16* wgA = wimg;                 // 128 x 256
  __bf16* wbA = wgA + CA_ * CS_;
  __bf16* wgT = wbA + CA_ * CS_;
  __bf16* wbT = wgT + CA_ * CS_;
  __bf16* wsg = wbT + CA_ * CS_;
  __bf16* wtc = wsg + CA_ * CS_;
  __bf16* wq  = wtc + CA_ * CS_;      // 128 x 128
  __bf16* wkv = wq + CA_ * CA_;       // 256 x 128
  __bf16* w1  = wkv + 2 * CA_ * CA_;  // 256 x 128
  __bf16* w2  = w1 + 2 * CA_ * CA_;   // 256 x 128
  __bf16* wo  = w2 + 2 * CA_ * CA_;   // 128 x 256

  // ---- one-time weight prep (tiny; stays hot in L2) ----
  const int g32k = (CS_ * CA_ + 255) / 256;   // 128 blocks
  const int g16k = (CA_ * CA_ + 255) / 256;   // 64 blocks
  k0_wprep<<<g32k, 256, 0, stream>>>(a_Wg, wgA, CS_, CA_);
  k0_wprep<<<g32k, 256, 0, stream>>>(a_Wb, wbA, CS_, CA_);
  k0_wprep<<<g32k, 256, 0, stream>>>(t_Wg, wgT, CS_, CA_);
  k0_wprep<<<g32k, 256, 0, stream>>>(t_Wb, wbT, CS_, CA_);
  k0_wprep<<<g32k, 256, 0, stream>>>(sg_W, wsg, CS_, CA_);
  k0_wprep<<<g32k, 256, 0, stream>>>(t_Wc, wtc, CS_, CA_);
  k0_wprep<<<g16k, 256, 0, stream>>>(q_W, wq, CA_, CA_);
  k0_wprep<<<g32k, 256, 0, stream>>>(kv_W, wkv, CA_, 2 * CA_);
  k0_wprep<<<g32k, 256, 0, stream>>>(t_W1, w1, CA_, 2 * CA_);
  k0_wprep<<<g32k, 256, 0, stream>>>(t_W2, w2, CA_, 2 * CA_);
  k0_wprep<<<g32k, 256, 0, stream>>>(t_Wo, wo, CS_, CA_);

  k1_res_cond<<<B_ * R_ / 16, 128, 0, stream>>>(
      resF, a_ln_w, t_ln_w, wgA, wbA, wgT, wbT, wsg, wtc, a_bg, t_bg, sg_b,
      t_bc, gateA, biasA, gateT, biasT, gS, gT);

  k2_atom_qkv<<<B_ * N_ / 16, 128, 0, stream>>>(
      atomF, ari, gateA, biasA, wq, q_b, wkv, qWs, kvWs);

  const size_t smem3 = (size_t)(WQ_ * CA_ + 2 * WK_ * CA_) * sizeof(__bf16) +
                       (size_t)(H_ * WQ_ * WK_) * sizeof(float) +
                       (size_t)(8 * 16 * WK_) * sizeof(__bf16);  // 172032 B
  k3_attn<<<B_ * NW_, 256, smem3, stream>>>(
      qWs, kvWs, apF, bij_ln_w, bij_ln_b, bij_W, ari, gS, amask, atomF,
      atomInter);

  k4_transition<<<B_ * N_ / 16, 128, 0, stream>>>(
      atomInter, ari, amask, gateT, biasT, gT, w1, w2, wo, out);
}
